// EmformerAttention_25666724561392
// MI455X (gfx1250) — compile-verified, matrix operand order checked by reading, not run on hardware
//
#include <hip/hip_runtime.h>

typedef __attribute__((ext_vector_type(16))) _Float16 v16h;
typedef __attribute__((ext_vector_type(8)))  _Float16 v8h;
typedef __attribute__((ext_vector_type(8)))  float    v8f;

namespace {
constexpr int Ud = 512, Bd = 16, Dd = 512, Hd = 8, Rd = 32, Sd = 16, Md = 32;
constexpr int HDd = 64;            // D / H
constexpr int Qd  = Rd + Ud + Sd;  // 560
constexpr int QPd = 576;           // Q padded to /64
constexpr int KVd = Md + Rd + Ud;  // 576
constexpr int PEd = 2 * Ud - 1;    // 1023
constexpr int PEPd = 1024;         // PE padded
constexpr int BHd = Bd * Hd;       // 128
constexpr float SCALEf = 0.125f;   // 64^-0.5
constexpr float NEGINF = -100000000.0f;
}

// ---------------- WMMA fragment helpers ----------------------------------
// 16-bit A/B fragment layout (ISA 7.12.2): lane<16 -> row M=lane, K chunks
// {0..7, 16..23}; lane>=16 -> row M=lane-16, K chunks {8..15, 24..31}.
__device__ __forceinline__ v16h load_frag(const _Float16* p, int ld, int lane) {
    const int row = lane & 15;
    const int kc  = (lane & 16) ? 8 : 0;
    const _Float16* q = p + (size_t)row * (size_t)ld + kc;
    v8h lo = *(const v8h*)(q);
    v8h hi = *(const v8h*)(q + 16);
    return __builtin_shufflevector(lo, hi, 0,1,2,3,4,5,6,7,8,9,10,11,12,13,14,15);
}

__device__ __forceinline__ v8f wmma_f16(v16h a, v16h b, v8f c) {
    return __builtin_amdgcn_wmma_f32_16x16x32_f16(false, a, false, b, (short)0, c,
                                                  false, false);
}

// One wave computes a 32x32 tile of A(MxK) * B(NxK)^T, K multiple of 32.
__device__ __forceinline__ void mm32x32(const _Float16* __restrict__ Abase,
                                        const _Float16* __restrict__ Bbase,
                                        int lda, int ldb, int K, int lane,
                                        v8f c[2][2]) {
    for (int k0 = 0; k0 < K; k0 += 32) {
        v16h a0 = load_frag(Abase + k0, lda, lane);
        v16h a1 = load_frag(Abase + (size_t)16 * lda + k0, lda, lane);
        v16h b0 = load_frag(Bbase + k0, ldb, lane);
        v16h b1 = load_frag(Bbase + (size_t)16 * ldb + k0, ldb, lane);
        if (k0 + 32 < K) {
            __builtin_prefetch(Abase + (size_t)(lane & 15) * lda + k0 + 32, 0, 0);
            __builtin_prefetch(Bbase + (size_t)(lane & 15) * ldb + k0 + 32, 0, 0);
        }
        c[0][0] = wmma_f16(a0, b0, c[0][0]);
        c[0][1] = wmma_f16(a0, b1, c[0][1]);
        c[1][0] = wmma_f16(a1, b0, c[1][0]);
        c[1][1] = wmma_f16(a1, b1, c[1][1]);
    }
}

// C/D layout: VGPR r, lanes 0-15 -> (M=r, N=lane); lanes 16-31 -> (M=8+r, N=lane-16)
#define EPILOGUE_LOOP(BODY)                                                     \
    {                                                                           \
        const int mo = (lane & 16) ? 8 : 0;                                     \
        const int nl = lane & 15;                                               \
        _Pragma("unroll") for (int i = 0; i < 2; ++i)                           \
        _Pragma("unroll") for (int j = 0; j < 2; ++j)                           \
        _Pragma("unroll") for (int r = 0; r < 8; ++r) {                         \
            const int m = tm + i * 16 + mo + r;                                 \
            const int n = tn + j * 16 + nl;                                     \
            const float v = c[i][j][r];                                         \
            BODY                                                                \
        }                                                                       \
    }

// ---------------- elementwise helpers -------------------------------------
__global__ void cvt16(const float* __restrict__ s, _Float16* __restrict__ d, int n) {
    int i = blockIdx.x * 256 + threadIdx.x;
    if (i < n) d[i] = (_Float16)s[i];
}

__global__ void cvt16pad(const float* __restrict__ s, _Float16* __restrict__ d,
                         int n, int ntot) {
    int i = blockIdx.x * 256 + threadIdx.x;
    if (i < ntot) d[i] = (i < n) ? (_Float16)s[i] : (_Float16)0.f;
}

// ---------------- projections: C = A * B^T (+bias) -------------------------
__global__ void gemm_nt_bias(const _Float16* __restrict__ A,
                             const _Float16* __restrict__ B,
                             const float* __restrict__ bias,
                             float* __restrict__ C,
                             int lda, int ldb, int ldc, int K) {
    const int lane = threadIdx.x & 31, w = threadIdx.x >> 5;
    const int tm = blockIdx.y * 64 + (w >> 1) * 32;
    const int tn = blockIdx.x * 64 + (w & 1) * 32;
    v8f c[2][2] = {};
    mm32x32(A + (size_t)tm * lda, B + (size_t)tn * ldb, lda, ldb, K, lane, c);
    EPILOGUE_LOOP({
        float o = v + (bias ? bias[n] : 0.f);
        C[(size_t)m * ldc + n] = o;
    })
}

// ---------------- attention prep -------------------------------------------
__global__ void prep_qu(const float* __restrict__ query, const float* __restrict__ pbu,
                        _Float16* __restrict__ qu) {
    int idx = blockIdx.x * 256 + threadIdx.x;
    if (idx >= BHd * QPd * HDd) return;
    int hd = idx & 63, q = (idx >> 6) % QPd, bh = idx / (QPd * HDd);
    int b = bh >> 3, h = bh & 7;
    float v = 0.f;
    if (q < Qd) v = query[((size_t)q * Bd + b) * Dd + h * HDd + hd] + pbu[h * HDd + hd];
    qu[idx] = (_Float16)v;
}

__global__ void prep_uv(const float* __restrict__ query, const float* __restrict__ pbv,
                        _Float16* __restrict__ uv) {
    int idx = blockIdx.x * 256 + threadIdx.x;
    if (idx >= BHd * Ud * HDd) return;
    int hd = idx & 63, u = (idx >> 6) % Ud, bh = idx / (Ud * HDd);
    int b = bh >> 3, h = bh & 7;
    float v = query[((size_t)(Rd + u) * Bd + b) * Dd + h * HDd + hd] + pbv[h * HDd + hd];
    uv[idx] = (_Float16)v;
}

__global__ void prep_kv(const float* __restrict__ kvf, _Float16* __restrict__ rk,
                        _Float16* __restrict__ rvT) {
    int idx = blockIdx.x * 256 + threadIdx.x;
    if (idx >= BHd * KVd * HDd) return;
    int hd = idx & 63, k = (idx >> 6) % KVd, bh = idx / (KVd * HDd);
    int b = bh >> 3, h = bh & 7;
    const float* row = kvf + ((size_t)k * Bd + b) * (2 * Dd);
    rk[idx] = (_Float16)row[h * HDd + hd];
    rvT[((size_t)bh * HDd + hd) * KVd + k] = (_Float16)row[Dd + h * HDd + hd];
}

__global__ void prep_pos(const float* __restrict__ posf, _Float16* __restrict__ posh) {
    int idx = blockIdx.x * 256 + threadIdx.x;
    if (idx >= Hd * PEPd * HDd) return;
    int hd = idx & 63, p = (idx >> 6) & (PEPd - 1), h = idx / (PEPd * HDd);
    float v = (p < PEd) ? posf[(size_t)p * Dd + h * HDd + hd] : 0.f;
    posh[idx] = (_Float16)v;
}

// ---------------- matrix_ac: qu * rk^T per (b,h) ----------------------------
__global__ void gemm_ac(const _Float16* __restrict__ qu, const _Float16* __restrict__ rk,
                        float* __restrict__ ac) {
    const int bh = blockIdx.z;
    const int lane = threadIdx.x & 31, w = threadIdx.x >> 5;
    const int tm = blockIdx.y * 64 + (w >> 1) * 32;
    const int tn = blockIdx.x * 64 + (w & 1) * 32;
    const _Float16* A = qu + (size_t)bh * QPd * HDd + (size_t)tm * HDd;
    const _Float16* B = rk + (size_t)bh * KVd * HDd + (size_t)tn * HDd;
    float* C = ac + (size_t)bh * QPd * KVd;
    v8f c[2][2] = {};
    mm32x32(A, B, HDd, HDd, HDd, lane, c);
    EPILOGUE_LOOP({ C[(size_t)m * KVd + n] = v; })
}

// ---------------- bd: uv * pos^T with fused rel-shift accumulation ----------
__global__ void gemm_bd(const _Float16* __restrict__ uv, const _Float16* __restrict__ posh,
                        float* __restrict__ ac) {
    const int bh = blockIdx.z;
    const int h = bh & 7;
    const int lane = threadIdx.x & 31, w = threadIdx.x >> 5;
    const int tm = blockIdx.y * 64 + (w >> 1) * 32;  // u
    const int tn = blockIdx.x * 64 + (w & 1) * 32;   // p
    const _Float16* A = uv + (size_t)bh * Ud * HDd + (size_t)tm * HDd;
    const _Float16* B = posh + (size_t)h * PEPd * HDd + (size_t)tn * HDd;
    v8f c[2][2] = {};
    mm32x32(A, B, HDd, HDd, HDd, lane, c);
    EPILOGUE_LOOP({
        const int jj = n - (Ud - 1) + m;  // rel-shift column
        if (jj >= 0 && jj < Ud)
            ac[((size_t)bh * QPd + (Rd + m)) * KVd + (Md + Rd) + jj] += v;
    })
}

// ---------------- masked softmax: one wave32 per row ------------------------
__global__ void softmax_row(const float* __restrict__ ac,
                            const unsigned char* __restrict__ am,
                            const int* __restrict__ lengths,
                            _Float16* __restrict__ probs) {
    const int q = blockIdx.x, bh = blockIdx.y, lane = threadIdx.x;
    _Float16* prow = probs + ((size_t)bh * QPd + q) * KVd;
    if (q >= Qd) {  // zero padded rows so downstream WMMA reads are clean
        #pragma unroll
        for (int i = 0; i < KVd / 32; ++i) prow[lane + i * 32] = (_Float16)0.f;
        return;
    }
    const float* arow = ac + ((size_t)bh * QPd + q) * KVd;
    const int b = bh >> 3;
    const int kcut = (KVd - Ud) + lengths[b];  // 64 + len
    float vals[KVd / 32];
    float mx = NEGINF;
    #pragma unroll
    for (int i = 0; i < KVd / 32; ++i) {
        const int k = lane + i * 32;
        float wv = arow[k] * SCALEf;
        if (am[(size_t)q * KVd + k]) wv = NEGINF;
        if (k >= kcut) wv = NEGINF;
        vals[i] = wv;
        mx = fmaxf(mx, wv);
    }
    #pragma unroll
    for (int off = 16; off > 0; off >>= 1) mx = fmaxf(mx, __shfl_xor(mx, off));
    float s = 0.f;
    #pragma unroll
    for (int i = 0; i < KVd / 32; ++i) {
        float e = __expf(vals[i] - mx);
        vals[i] = e;
        s += e;
    }
    #pragma unroll
    for (int off = 16; off > 0; off >>= 1) s += __shfl_xor(s, off);
    const float inv = 1.f / s;
    #pragma unroll
    for (int i = 0; i < KVd / 32; ++i) prow[lane + i * 32] = (_Float16)(vals[i] * inv);
}

// ---------------- attn: probs * rv, scatter to (q*B+b, h*HD+hd) layout ------
__global__ void gemm_attn(const _Float16* __restrict__ probs,
                          const _Float16* __restrict__ rvT,
                          _Float16* __restrict__ attnX) {
    const int bh = blockIdx.z;
    const int b = bh >> 3, h = bh & 7;
    const int lane = threadIdx.x & 31, w = threadIdx.x >> 5;
    const int tm = blockIdx.y * 64 + (w >> 1) * 32;  // q
    const int tn = blockIdx.x * 64 + (w & 1) * 32;   // hd
    const _Float16* A = probs + (size_t)bh * QPd * KVd + (size_t)tm * KVd;
    const _Float16* B = rvT + (size_t)bh * HDd * KVd + (size_t)tn * KVd;
    v8f c[2][2] = {};
    mm32x32(A, B, KVd, KVd, KVd, lane, c);
    EPILOGUE_LOOP({
        if (m < Qd)
            attnX[((size_t)m * Bd + b) * Dd + h * HDd + n] = (_Float16)v;
    })
}

// ---------------- output projection + clip + split --------------------------
__global__ void gemm_out(const _Float16* __restrict__ attnX,
                         const _Float16* __restrict__ Wo16,
                         const float* __restrict__ bo,
                         float* __restrict__ dout) {
    const int lane = threadIdx.x & 31, w = threadIdx.x >> 5;
    const int tm = blockIdx.y * 64 + (w >> 1) * 32;
    const int tn = blockIdx.x * 64 + (w & 1) * 32;
    v8f c[2][2] = {};
    mm32x32(attnX + (size_t)tm * Dd, Wo16 + (size_t)tn * Dd, Dd, Dd, Dd, lane, c);
    EPILOGUE_LOOP({
        float o = v + bo[n];
        if (m >= (Rd + Ud) * Bd) o = fminf(10.f, fmaxf(-10.f, o));  // summary rows
        dout[(size_t)m * Dd + n] = o;
    })
}

// ---------------- key_t / val_t outputs -------------------------------------
__global__ void kv_out(const float* __restrict__ kvf, float* __restrict__ dout) {
    size_t idx = (size_t)blockIdx.x * 256 + threadIdx.x;
    const size_t total = (size_t)KVd * Bd * Dd;  // 4718592
    if (idx >= total) return;
    const int d = (int)(idx & (Dd - 1));
    const size_t n = idx >> 9;
    const size_t base = (size_t)Qd * Bd * Dd;  // 4587520
    dout[base + idx]         = kvf[n * (2 * Dd) + d];
    dout[base + total + idx] = kvf[n * (2 * Dd) + Dd + d];
}

// ---------------- host side --------------------------------------------------
extern "C" void kernel_launch(void* const* d_in, const int* in_sizes, int n_in,
                              void* d_out, int out_size, void* d_ws, size_t ws_size,
                              hipStream_t stream) {
    const float* utt  = (const float*)d_in[0];
    const int*   lens = (const int*)d_in[1];
    const float* rc   = (const float*)d_in[2];
    const float* summ = (const float*)d_in[3];
    const float* mem  = (const float*)d_in[4];
    const unsigned char* amask = (const unsigned char*)d_in[5];
    const float* pemb = (const float*)d_in[6];
    const float* Wq   = (const float*)d_in[7];
    const float* bq   = (const float*)d_in[8];
    const float* Wkv  = (const float*)d_in[9];
    const float* bkv  = (const float*)d_in[10];
    const float* Wo   = (const float*)d_in[11];
    const float* bo   = (const float*)d_in[12];
    const float* Wpos = (const float*)d_in[13];
    const float* pbu  = (const float*)d_in[14];
    const float* pbv  = (const float*)d_in[15];
    float* dout = (float*)d_out;

    // workspace carve-up (256B aligned)
    char* ws = (char*)d_ws;
    size_t off = 0;
    auto take = [&](size_t bytes) -> char* {
        off = (off + 255) & ~(size_t)255;
        char* p = ws + off;
        off += bytes;
        return p;
    };
    const size_t Nq = (size_t)Qd * Bd;    // 8960 rows
    const size_t Nk = (size_t)KVd * Bd;   // 9216 rows
    _Float16* Xq16   = (_Float16*)take(Nq * Dd * 2);
    _Float16* Xkv16  = (_Float16*)take(Nk * Dd * 2);
    _Float16* Wq16   = (_Float16*)take((size_t)Dd * Dd * 2);
    _Float16* Wkv16  = (_Float16*)take((size_t)2 * Dd * Dd * 2);
    _Float16* Wo16   = (_Float16*)take((size_t)Dd * Dd * 2);
    _Float16* Wpos16 = (_Float16*)take((size_t)Dd * Dd * 2);
    _Float16* pe16   = (_Float16*)take((size_t)PEPd * Dd * 2);
    float*    queryf = (float*)take(Nq * Dd * 4);
    float*    kvf    = (float*)take(Nk * 2 * Dd * 4);
    float*    posf   = (float*)take((size_t)PEPd * Dd * 4);
    _Float16* qu16   = (_Float16*)take((size_t)BHd * QPd * HDd * 2);
    _Float16* uv16   = (_Float16*)take((size_t)BHd * Ud * HDd * 2);
    _Float16* rk16   = (_Float16*)take((size_t)BHd * KVd * HDd * 2);
    _Float16* rvT16  = (_Float16*)take((size_t)BHd * HDd * KVd * 2);
    _Float16* posh16 = (_Float16*)take((size_t)Hd * PEPd * HDd * 2);
    float*    acf    = (float*)take((size_t)BHd * QPd * KVd * 4);
    _Float16* probs16= (_Float16*)take((size_t)BHd * QPd * KVd * 2);
    _Float16* attnX16= (_Float16*)take(Nq * Dd * 2);
    (void)ws_size; (void)n_in; (void)in_sizes; (void)out_size;

    auto grid1 = [](size_t n) { return dim3((unsigned)((n + 255) / 256)); };
    const size_t nRC = (size_t)Rd * Bd * Dd;   // 262144
    const size_t nU  = (size_t)Ud * Bd * Dd;   // 4194304
    const size_t nS  = (size_t)Sd * Bd * Dd;   // 131072
    const size_t nM  = (size_t)Md * Bd * Dd;   // 262144

    // 1) f32 -> f16 staging (concat along axis 0 == flat concat)
    cvt16<<<grid1(nRC), 256, 0, stream>>>(rc,   Xq16,              (int)nRC);
    cvt16<<<grid1(nU),  256, 0, stream>>>(utt,  Xq16 + nRC,        (int)nU);
    cvt16<<<grid1(nS),  256, 0, stream>>>(summ, Xq16 + nRC + nU,   (int)nS);
    cvt16<<<grid1(nM),  256, 0, stream>>>(mem,  Xkv16,             (int)nM);
    cvt16<<<grid1(nRC), 256, 0, stream>>>(rc,   Xkv16 + nM,        (int)nRC);
    cvt16<<<grid1(nU),  256, 0, stream>>>(utt,  Xkv16 + nM + nRC,  (int)nU);
    cvt16<<<grid1(Dd*Dd),   256, 0, stream>>>(Wq,   Wq16,   Dd * Dd);
    cvt16<<<grid1(2*Dd*Dd), 256, 0, stream>>>(Wkv,  Wkv16,  2 * Dd * Dd);
    cvt16<<<grid1(Dd*Dd),   256, 0, stream>>>(Wo,   Wo16,   Dd * Dd);
    cvt16<<<grid1(Dd*Dd),   256, 0, stream>>>(Wpos, Wpos16, Dd * Dd);
    cvt16pad<<<grid1((size_t)PEPd*Dd), 256, 0, stream>>>(pemb, pe16, PEd * Dd, PEPd * Dd);

    // 2) projections (WMMA f16, f32 accumulate)
    gemm_nt_bias<<<dim3(Dd/64,     (unsigned)(Nq/64)), 128, 0, stream>>>(
        Xq16, Wq16, bq, queryf, Dd, Dd, Dd, Dd);
    gemm_nt_bias<<<dim3(2*Dd/64,   (unsigned)(Nk/64)), 128, 0, stream>>>(
        Xkv16, Wkv16, bkv, kvf, Dd, Dd, 2 * Dd, Dd);
    gemm_nt_bias<<<dim3(Dd/64,     PEPd/64), 128, 0, stream>>>(
        pe16, Wpos16, nullptr, posf, Dd, Dd, Dd, Dd);

    // 3) per-head staging
    prep_qu<<<grid1((size_t)BHd*QPd*HDd), 256, 0, stream>>>(queryf, pbu, qu16);
    prep_uv<<<grid1((size_t)BHd*Ud*HDd),  256, 0, stream>>>(queryf, pbv, uv16);
    prep_kv<<<grid1((size_t)BHd*KVd*HDd), 256, 0, stream>>>(kvf, rk16, rvT16);
    prep_pos<<<grid1((size_t)Hd*PEPd*HDd),256, 0, stream>>>(posf, posh16);

    // 4) matrix_ac, then bd with fused rel-shift accumulation
    gemm_ac<<<dim3(KVd/64, QPd/64, BHd), 128, 0, stream>>>(qu16, rk16, acf);
    gemm_bd<<<dim3(PEPd/64, Ud/64, BHd), 128, 0, stream>>>(uv16, posh16, acf);

    // 5) masked softmax (wave32 per row)
    softmax_row<<<dim3(QPd, BHd), 32, 0, stream>>>(acf, amask, lens, probs16);

    // 6) attention * V, scattered into (q*B+b, h*HD+hd) layout
    gemm_attn<<<dim3(HDd/64, QPd/64, BHd), 128, 0, stream>>>(probs16, rvT16, attnX16);

    // 7) output projection (+bias, clip summary rows) straight into d_out
    gemm_out<<<dim3(Dd/64, (unsigned)(Nq/64)), 128, 0, stream>>>(attnX16, Wo16, bo, dout);

    // 8) key_t / val_t outputs
    kv_out<<<grid1((size_t)KVd*Bd*Dd), 256, 0, stream>>>(kvf, dout);
}
